// HeteroMessagePassingLayer_11373073400378
// MI455X (gfx1250) — matrix operand order repaired and checked by reading, not compiled, not run on hardware
//
#include <hip/hip_runtime.h>
#include <hip/hip_bf16.h>
#include <math.h>

// ---------------------------------------------------------------------------
// HeteroConv(GCN glom->glom, GATv2 imm->glom, GIN glom->imm) + LayerNorm+ReLU
// MI455X / gfx1250, wave32.  GEMMs use v_wmma_f32_16x16x32_f16.
// Working set (~125 MB) fits in 192 MB L2 -> edge atomics stay on-chip.
// ---------------------------------------------------------------------------

#define DFEAT 256      // feature dim (in == out)
#define KCHUNK 64      // K-chunk staged in LDS per GEMM iteration
#define BS_STRIDE 72   // padded LDS stride (halfs) to reduce bank conflicts

typedef __attribute__((ext_vector_type(16))) _Float16 v16h;
typedef __attribute__((ext_vector_type(8)))  float    v8f;
typedef __attribute__((ext_vector_type(2)))  _Float16 v2h;

// ---------------------------------------------------------------------------
// WMMA GEMM: C[nrows x 256] = A[nrows x 256] * B[256 x 256] (+ bias)
// 256 threads = 8 waves / block. Each wave computes a 16-row strip across all
// 16 column tiles (16 x v8f accumulators). B is staged to LDS as f16 in
// K-chunks of 64 (transposed, padded) and reused by all 8 waves.
// Fragment layouts per CDNA5 ISA 7.12.2 (wave32):
//   A 16x32 f16 : lane L -> row M=L&15, half h=L>>4;
//                 vgpr v<4 holds K=2v+8h,+1 ; v>=4 holds K=16+2(v-4)+8h,+1
//   B 32x16 f16 : lane L -> col N=L&15, half h=L>>4; vgpr v holds same K as A
//   C 16x16 f32 : vgpr v -> row v+8h, col L&15
// ---------------------------------------------------------------------------
__global__ __launch_bounds__(256)
void gemm_wmma_f16(const float* __restrict__ A, const float* __restrict__ B,
                   const float* __restrict__ bias, float* __restrict__ C,
                   int nrows)
{
    __shared__ _Float16 Bs[DFEAT * BS_STRIDE];   // Bs[col][k_local], 36 KB

    const int tid     = threadIdx.x;
    const int wave    = tid >> 5;
    const int lane    = tid & 31;
    const int laneMod = lane & 15;
    const int laneHi  = lane >> 4;

    const int rowTile = blockIdx.x * 8 + wave;
    const int rowBase = rowTile * 16;
    const bool active = rowBase < nrows;

    v8f acc[16];
#pragma unroll
    for (int ct = 0; ct < 16; ++ct)
#pragma unroll
        for (int v = 0; v < 8; ++v) acc[ct][v] = 0.0f;

    for (int kc = 0; kc < DFEAT / KCHUNK; ++kc) {
        // Cooperative load of B chunk: coalesced global reads (col = tid),
        // transposed f16 store into LDS.
#pragma unroll 8
        for (int i = 0; i < KCHUNK; ++i)
            Bs[tid * BS_STRIDE + i] =
                (_Float16)B[(size_t)(kc * KCHUNK + i) * DFEAT + tid];
        __syncthreads();

        if (active) {
            const float* arow =
                A + (size_t)(rowBase + laneMod) * DFEAT + kc * KCHUNK;
#pragma unroll
            for (int kt = 0; kt < 2; ++kt) {           // two 32-K WMMA steps
                v16h a;
#pragma unroll
                for (int v = 0; v < 8; ++v) {
                    int kb = (v < 4) ? (2 * v + 8 * laneHi)
                                     : (16 + 2 * (v - 4) + 8 * laneHi);
                    kb += kt * 32;
                    float2 f2 = *reinterpret_cast<const float2*>(arow + kb);
                    a[2 * v]     = (_Float16)f2.x;
                    a[2 * v + 1] = (_Float16)f2.y;
                }
#pragma unroll
                for (int ct = 0; ct < 16; ++ct) {
                    const int col = ct * 16 + laneMod;
                    v16h b;
#pragma unroll
                    for (int v = 0; v < 8; ++v) {
                        const int kl = kt * 32 + ((v < 4) ? (2 * v + 8 * laneHi)
                                                          : (16 + 2 * (v - 4) + 8 * laneHi));
                        v2h p = *reinterpret_cast<const v2h*>(
                            &Bs[col * BS_STRIDE + kl]);
                        b[2 * v]     = p.x;
                        b[2 * v + 1] = p.y;
                    }
                    acc[ct] = __builtin_amdgcn_wmma_f32_16x16x32_f16(
                        false, a, false, b, (short)0, acc[ct], false, false);
                }
            }
        }
        __syncthreads();
    }

    if (active) {
#pragma unroll
        for (int ct = 0; ct < 16; ++ct) {
            const int col  = ct * 16 + laneMod;
            const float bv = bias ? bias[col] : 0.0f;
#pragma unroll
            for (int v = 0; v < 8; ++v) {
                const int row = rowBase + v + 8 * laneHi;
                C[(size_t)row * DFEAT + col] = acc[ct][v] + bv;
            }
        }
    }
}

// ---------------------------------------------------------------------------
// Helpers / elementwise & edge kernels
// ---------------------------------------------------------------------------
__device__ inline void atomicMaxFloat(float* addr, float val)
{
    unsigned int* ua  = (unsigned int*)addr;
    unsigned int  old = *ua;
    while (__uint_as_float(old) < val) {
        unsigned int assumed = old;
        old = atomicCAS(ua, assumed, __float_as_uint(val));
        if (old == assumed) break;
    }
}

// deg=1 (self loop), m=-inf, denom=0
__global__ void node_init_kernel(float* deg, float* mmax, float* denom, int n)
{
    int i = blockIdx.x * blockDim.x + threadIdx.x;
    if (i < n) { deg[i] = 1.0f; mmax[i] = -__builtin_inff(); denom[i] = 0.0f; }
}

__global__ void deg_scatter_kernel(const int* __restrict__ ei,
                                   const float* __restrict__ ew,
                                   float* deg, int ne)
{
    int e = blockIdx.x * blockDim.x + threadIdx.x;
    if (e < ne) atomicAdd(&deg[ei[ne + e]], ew[e]);
}

__global__ void dinv_kernel(const float* deg, float* dinv, int n)
{
    int i = blockIdx.x * blockDim.x + threadIdx.x;
    if (i < n) dinv[i] = rsqrtf(deg[i]);
}

// acc_glom = dinv^2 * xw + b_gcn + b_gat   (GCN self-loop term + both biases)
__global__ void glom_init_kernel(const float* __restrict__ dinv,
                                 const float* __restrict__ xw,
                                 const float* __restrict__ b_gcn,
                                 const float* __restrict__ b_gat,
                                 float* acc, int n)
{
    long idx = (long)blockIdx.x * blockDim.x + threadIdx.x;
    if (idx < (long)n * DFEAT) {
        int i = (int)(idx >> 8), j = (int)(idx & 255);
        acc[idx] = dinv[i] * dinv[i] * xw[idx] + b_gcn[j] + b_gat[j];
    }
}

// 64 threads per edge, float4 per thread: acc[dst] += coef * xw[src]
__global__ void gcn_scatter_kernel(const int* __restrict__ ei,
                                   const float* __restrict__ ew,
                                   const float* __restrict__ dinv,
                                   const float* __restrict__ xw,
                                   float* acc, int ne)
{
    long tid = (long)blockIdx.x * blockDim.x + threadIdx.x;
    int e = (int)(tid >> 6);
    if (e >= ne) return;
    int j0 = (int)(tid & 63) << 2;
    int s = ei[e], d = ei[ne + e];
    float coef = dinv[s] * ew[e] * dinv[d];
    float4 x = *reinterpret_cast<const float4*>(xw + (size_t)s * DFEAT + j0);
    float* o = acc + (size_t)d * DFEAT + j0;
    atomicAdd(o + 0, coef * x.x);
    atomicAdd(o + 1, coef * x.y);
    atomicAdd(o + 2, coef * x.z);
    atomicAdd(o + 3, coef * x.w);
}

// one wave per edge: logit = att . leaky(hl[s]+hr[d]+ea*We) ; segment max
__global__ __launch_bounds__(256)
void gat_logits_kernel(const int* __restrict__ ei, const float* __restrict__ ea,
                       const float* __restrict__ hl, const float* __restrict__ hr,
                       const float* __restrict__ We, const float* __restrict__ att,
                       float* logits, float* mmax, int ne)
{
    long gid = (long)blockIdx.x * blockDim.x + threadIdx.x;
    int e = (int)(gid >> 5);
    if (e >= ne) return;
    int lane = threadIdx.x & 31;
    int s = ei[e], d = ei[ne + e];
    float eav = ea[e];
    const float* ls = hl + (size_t)s * DFEAT;
    const float* rd = hr + (size_t)d * DFEAT;
    float accv = 0.0f;
#pragma unroll
    for (int i = 0; i < 8; ++i) {
        int j = i * 32 + lane;
        float z = ls[j] + rd[j] + eav * We[j];
        z = (z > 0.0f) ? z : 0.2f * z;
        accv += z * att[j];
    }
#pragma unroll
    for (int off = 16; off; off >>= 1) accv += __shfl_xor(accv, off, 32);
    if (lane == 0) { logits[e] = accv; atomicMaxFloat(&mmax[d], accv); }
}

__global__ void gat_exp_kernel(const int* __restrict__ ei,
                               const float* __restrict__ mmax,
                               float* logits, float* denom, int ne)
{
    int e = blockIdx.x * blockDim.x + threadIdx.x;
    if (e >= ne) return;
    int d = ei[ne + e];
    float ex = expf(logits[e] - mmax[d]);
    logits[e] = ex;
    atomicAdd(&denom[d], ex);
}

__global__ void gat_scatter_kernel(const int* __restrict__ ei,
                                   const float* __restrict__ logits,
                                   const float* __restrict__ denom,
                                   const float* __restrict__ hl,
                                   float* acc, int ne)
{
    long tid = (long)blockIdx.x * blockDim.x + threadIdx.x;
    int e = (int)(tid >> 6);
    if (e >= ne) return;
    int j0 = (int)(tid & 63) << 2;
    int s = ei[e], d = ei[ne + e];
    float alpha = logits[e] / denom[d];
    float4 h = *reinterpret_cast<const float4*>(hl + (size_t)s * DFEAT + j0);
    float* o = acc + (size_t)d * DFEAT + j0;
    atomicAdd(o + 0, alpha * h.x);
    atomicAdd(o + 1, alpha * h.y);
    atomicAdd(o + 2, alpha * h.z);
    atomicAdd(o + 3, alpha * h.w);
}

// tmp = (1+eps)*x_imm
__global__ void gin_init_kernel(const float* __restrict__ x_imm,
                                const float* __restrict__ eps,
                                float* tmp, int n)
{
    long idx = (long)blockIdx.x * blockDim.x + threadIdx.x;
    if (idx < (long)n * DFEAT) tmp[idx] = (1.0f + eps[0]) * x_imm[idx];
}

// tmp[dst] += x_glom[src]
__global__ void gin_scatter_kernel(const int* __restrict__ ei,
                                   const float* __restrict__ xg,
                                   float* tmp, int ne)
{
    long tid = (long)blockIdx.x * blockDim.x + threadIdx.x;
    int e = (int)(tid >> 6);
    if (e >= ne) return;
    int j0 = (int)(tid & 63) << 2;
    int s = ei[e], d = ei[ne + e];
    float4 x = *reinterpret_cast<const float4*>(xg + (size_t)s * DFEAT + j0);
    float* o = tmp + (size_t)d * DFEAT + j0;
    atomicAdd(o + 0, x.x);
    atomicAdd(o + 1, x.y);
    atomicAdd(o + 2, x.z);
    atomicAdd(o + 3, x.w);
}

// one wave per row: LayerNorm + ReLU
__global__ __launch_bounds__(256)
void ln_relu_kernel(const float* __restrict__ in, const float* __restrict__ gamma,
                    const float* __restrict__ beta, float* __restrict__ out, int n)
{
    long gid = (long)blockIdx.x * blockDim.x + threadIdx.x;
    int row = (int)(gid >> 5);
    if (row >= n) return;
    int lane = threadIdx.x & 31;
    const float* x = in + (size_t)row * DFEAT;
    float v[8];
    float s = 0.0f;
#pragma unroll
    for (int i = 0; i < 8; ++i) { v[i] = x[i * 32 + lane]; s += v[i]; }
#pragma unroll
    for (int off = 16; off; off >>= 1) s += __shfl_xor(s, off, 32);
    float mu = s * (1.0f / DFEAT);
    float vs = 0.0f;
#pragma unroll
    for (int i = 0; i < 8; ++i) { float d = v[i] - mu; vs += d * d; }
#pragma unroll
    for (int off = 16; off; off >>= 1) vs += __shfl_xor(vs, off, 32);
    float inv = rsqrtf(vs * (1.0f / DFEAT) + 1e-5f);
    float* o = out + (size_t)row * DFEAT;
#pragma unroll
    for (int i = 0; i < 8; ++i) {
        int j = i * 32 + lane;
        float y = (v[i] - mu) * inv * gamma[j] + beta[j];
        o[j] = y > 0.0f ? y : 0.0f;
    }
}

// ---------------------------------------------------------------------------
// Launch
// ---------------------------------------------------------------------------
extern "C" void kernel_launch(void* const* d_in, const int* in_sizes, int n_in,
                              void* d_out, int out_size, void* d_ws, size_t ws_size,
                              hipStream_t stream)
{
    const float* x_glom = (const float*)d_in[0];
    const float* x_imm  = (const float*)d_in[1];
    const float* ew_gg  = (const float*)d_in[2];
    const float* ea_ig  = (const float*)d_in[3];
    const float* W_gcn  = (const float*)d_in[4];
    const float* b_gcn  = (const float*)d_in[5];
    const float* Wl     = (const float*)d_in[6];
    const float* bl     = (const float*)d_in[7];
    const float* Wr     = (const float*)d_in[8];
    const float* br     = (const float*)d_in[9];
    const float* att    = (const float*)d_in[10];
    const float* We     = (const float*)d_in[11];
    const float* b_gat  = (const float*)d_in[12];
    const float* eps    = (const float*)d_in[13];
    const float* W_gin  = (const float*)d_in[14];
    const float* b_gin  = (const float*)d_in[15];
    const float* gamma  = (const float*)d_in[16];
    const float* beta   = (const float*)d_in[17];
    const int*   ei_gg  = (const int*)d_in[18];
    const int*   ei_ig  = (const int*)d_in[19];
    const int*   ei_gi  = (const int*)d_in[20];

    const int Nn = in_sizes[0] / DFEAT;   // 20000
    const int Ee = in_sizes[2];           // 320000
    const size_t ND = (size_t)Nn * DFEAT;

    // workspace layout (floats); total ~125 MB, fits MI455X's 192 MB L2
    float* ws     = (float*)d_ws;
    float* xw     = ws;             // [N,D] x_glom @ W_gcn
    float* hl     = ws + 1 * ND;    // [N,D]
    float* hr     = ws + 2 * ND;    // [N,D]
    float* accg   = ws + 3 * ND;    // [N,D] out_gcn + out_gat accumulator
    float* tmp    = ws + 4 * ND;    // [N,D] GIN pre-GEMM input
    float* ging   = ws + 5 * ND;    // [N,D] GIN GEMM output
    float* deg    = ws + 6 * ND;    // [N]
    float* dinv   = deg + Nn;       // [N]
    float* mmax   = dinv + Nn;      // [N]
    float* denom  = mmax + Nn;      // [N]
    float* logits = denom + Nn;     // [E]

    float* out_glom = (float*)d_out;
    float* out_imm  = (float*)d_out + ND;

    const int rowTiles   = (Nn + 15) / 16;
    const int gemmBlocks = (rowTiles + 7) / 8;
    const int ndBlocks   = (int)((ND + 255) / 256);
    const int nBlocks    = (Nn + 255) / 256;
    const int eBlocks    = (Ee + 255) / 256;
    const int eWaveBlks  = (int)(((long)Ee * 32 + 255) / 256);
    const int eScatBlks  = (int)(((long)Ee * 64 + 255) / 256);
    const int rowWaveBlk = (int)(((long)Nn * 32 + 255) / 256);

    // --- dense GEMMs (WMMA) ---
    gemm_wmma_f16<<<gemmBlocks, 256, 0, stream>>>(x_glom, W_gcn, nullptr, xw, Nn);
    gemm_wmma_f16<<<gemmBlocks, 256, 0, stream>>>(x_imm,  Wl,    bl,      hl, Nn);
    gemm_wmma_f16<<<gemmBlocks, 256, 0, stream>>>(x_glom, Wr,    br,      hr, Nn);

    // --- GCN (glom->glom) ---
    node_init_kernel<<<nBlocks, 256, 0, stream>>>(deg, mmax, denom, Nn);
    deg_scatter_kernel<<<eBlocks, 256, 0, stream>>>(ei_gg, ew_gg, deg, Ee);
    dinv_kernel<<<nBlocks, 256, 0, stream>>>(deg, dinv, Nn);
    glom_init_kernel<<<ndBlocks, 256, 0, stream>>>(dinv, xw, b_gcn, b_gat, accg, Nn);
    gcn_scatter_kernel<<<eScatBlks, 256, 0, stream>>>(ei_gg, ew_gg, dinv, xw, accg, Ee);

    // --- GATv2 (imm->glom) ---
    gat_logits_kernel<<<eWaveBlks, 256, 0, stream>>>(ei_ig, ea_ig, hl, hr, We, att,
                                                     logits, mmax, Ee);
    gat_exp_kernel<<<eBlocks, 256, 0, stream>>>(ei_ig, mmax, logits, denom, Ee);
    gat_scatter_kernel<<<eScatBlks, 256, 0, stream>>>(ei_ig, logits, denom, hl, accg, Ee);

    // --- GIN (glom->imm) ---
    gin_init_kernel<<<ndBlocks, 256, 0, stream>>>(x_imm, eps, tmp, Nn);
    gin_scatter_kernel<<<eScatBlks, 256, 0, stream>>>(ei_gi, x_glom, tmp, Ee);
    gemm_wmma_f16<<<gemmBlocks, 256, 0, stream>>>(tmp, W_gin, b_gin, ging, Nn);

    // --- shared LayerNorm + ReLU ---
    ln_relu_kernel<<<rowWaveBlk, 256, 0, stream>>>(accg, gamma, beta, out_glom, Nn);
    ln_relu_kernel<<<rowWaveBlk, 256, 0, stream>>>(ging, gamma, beta, out_imm, Nn);
}
